// GravNetRegressor_55241869361464
// MI455X (gfx1250) — matrix operand order repaired and checked by reading, not compiled, not run on hardware
//
#include <hip/hip_runtime.h>
#include <hip/hip_bf16.h>

// ---------------------------------------------------------------------------
// GravNet regressor for MI455X (gfx1250, wave32, WMMA).
// Dense layers: v_wmma_f32_16x16x32_f16, B fragments staged f16 in LDS
// (zero-padded, transposed -> two 16B LDS reads per lane),
// A fragments as vectorized float4 chunk loads + pack-convert.
// kNN: register sorted-insert top-12 + wave butterfly merge.
// ---------------------------------------------------------------------------

typedef __attribute__((ext_vector_type(16))) _Float16 v16h;
typedef __attribute__((ext_vector_type(8)))  float    v8f;

#define NTOT   8192
#define HID    64
#define SDIM   4
#define PDIM   16
#define KNN    12
#define GSZ    1024   // nodes per graph

// ----- fragment helpers (ISA 7.12.2 layouts, wave32) ------------------------
__device__ __forceinline__ void pack8(v16h& a, int base, float4 p0, float4 p1) {
  a[base + 0] = (_Float16)p0.x; a[base + 1] = (_Float16)p0.y;
  a[base + 2] = (_Float16)p0.z; a[base + 3] = (_Float16)p0.w;
  a[base + 4] = (_Float16)p1.x; a[base + 5] = (_Float16)p1.y;
  a[base + 6] = (_Float16)p1.z; a[base + 7] = (_Float16)p1.w;
}

// A: 16x32 f16 tile slice; lane's fragment = two contiguous 8-float runs.
__device__ __forceinline__ v16h load_a_frag(const float* __restrict__ A, int lda,
                                            int row, int k0, int K) {
  const int half = (threadIdx.x >> 4) & 1;
  const float* r = A + row * lda;
  v16h a;
  {
    const float* p = r + k0 + 8 * half;
    pack8(a, 0, ((const float4*)p)[0], ((const float4*)p)[1]);
  }
  if (k0 + 32 <= K) {                 // uniform, constant-folds after inlining
    const float* p = r + k0 + 16 + 8 * half;
    pack8(a, 8, ((const float4*)p)[0], ((const float4*)p)[1]);
  } else {
#pragma unroll
    for (int i = 8; i < 16; ++i) a[i] = (_Float16)0.f;
  }
  return a;
}

// A rows from concat [h(64) | agg(32)]; each 8-float chunk lies in one segment.
__device__ __forceinline__ const float* cseg(const float* __restrict__ h,
                                             const float* __restrict__ agg,
                                             int row, int s) {
  return (s < 64) ? (h + row * 64 + s) : (agg + row * 32 + (s - 64));
}
__device__ __forceinline__ v16h load_a_concat(const float* __restrict__ h,
                                              const float* __restrict__ agg,
                                              int row, int k0) {
  const int half = (threadIdx.x >> 4) & 1;
  v16h a;
  {
    const float* p = cseg(h, agg, row, k0 + 8 * half);
    pack8(a, 0, ((const float4*)p)[0], ((const float4*)p)[1]);
  }
  {
    const float* p = cseg(h, agg, row, k0 + 16 + 8 * half);
    pack8(a, 8, ((const float4*)p)[0], ((const float4*)p)[1]);
  }
  return a;
}

// B fragment: Wt is LDS, laid out [col][kp] f16, zero-padded; lane reads
// 8 f16 (16B, aligned) at k0 + 8*half and 8 f16 at k0 + 16 + 8*half.
__device__ __forceinline__ v16h load_b_lds(const _Float16* Wt, int kp, int col, int k0) {
  typedef __attribute__((ext_vector_type(8))) _Float16 v8h;
  const int half = (threadIdx.x >> 4) & 1;
  const _Float16* p = Wt + col * kp + k0 + 8 * half;
  const v8h lo = *(const v8h*)p;
  const v8h hi = *(const v8h*)(p + 16);
  return __builtin_shufflevector(lo, hi, 0, 1, 2, 3, 4, 5, 6, 7, 8, 9, 10, 11, 12, 13, 14, 15);
}

__device__ __forceinline__ v8f wmma_acc(v16h a, v16h b, v8f c) {
  return __builtin_amdgcn_wmma_f32_16x16x32_f16(false, a, false, b,
                                                (short)0, c, false, false);
}

// generic 16x16 tile: A (f32, row-major) x staged-LDS B, K in {32-padded,64,96}
__device__ __forceinline__ v8f wmma_tile(const float* __restrict__ A, int lda, int rowA,
                                         const _Float16* Wt, int kp, int col, int K) {
  v8f acc = {};
#pragma unroll
  for (int k0 = 0; k0 < 96; k0 += 32) {
    if (k0 >= K) break;
    acc = wmma_acc(load_a_frag(A, lda, rowA, k0, K), load_b_lds(Wt, kp, col, k0), acc);
  }
  return acc;
}

// ----- Kernel 1: h0 = relu(x @ Wi1 + b1) @ Wi2 + b2 -------------------------
// 64 blocks x 256 thr; wave handles 16 rows, full 64 cols; stage1 in LDS.
__global__ void k_in_mlp(const float* __restrict__ x,
                         const float* __restrict__ W1, const float* __restrict__ b1,
                         const float* __restrict__ W2, const float* __restrict__ b2,
                         float* __restrict__ h0) {
  __shared__ alignas(32) _Float16 Wt1[64 * 32];   // Wi1^T, K=16 padded to 32
  __shared__ alignas(32) _Float16 Wt2[64 * 64];   // Wi2^T
  __shared__ alignas(16) float t_sh[8][16 * 64];
  const int tid = threadIdx.x;
  const int lane = tid & 31, wave = tid >> 5, half = lane >> 4;
  const int colL = lane & 15;
  const int m0 = blockIdx.x * 128 + wave * 16;
  const int rowA = m0 + colL;
  __builtin_prefetch(W1, 0, 3);
  __builtin_prefetch(W2, 0, 3);
  for (int i = tid; i < 64 * 32; i += 256) {
    const int col = i >> 5, k = i & 31;
    Wt1[i] = (k < 16) ? (_Float16)W1[k * 64 + col] : (_Float16)0.f;
  }
  for (int i = tid; i < 64 * 64; i += 256) {
    const int col = i >> 6, k = i & 63;
    Wt2[i] = (_Float16)W2[k * 64 + col];
  }
  __syncthreads();
  // stage 1: [16,16] @ [16,64] -> LDS, relu
#pragma unroll
  for (int nt = 0; nt < 4; ++nt) {
    const int col = nt * 16 + colL;
    v8f acc = wmma_tile(x, 16, rowA, Wt1, 32, col, 32);  // K padded to 32
    const float bv = b1[col];
#pragma unroll
    for (int r = 0; r < 8; ++r)
      t_sh[wave][(r + 8 * half) * 64 + col] = fmaxf(acc[r] + bv, 0.f);
  }
  // stage 2: [16,64] @ [64,64] -> global (no relu)
#pragma unroll
  for (int nt = 0; nt < 4; ++nt) {
    const int col = nt * 16 + colL;
    v8f acc = wmma_tile(&t_sh[wave][0], 64, colL, Wt2, 64, col, 64);
    const float bv = b2[col];
#pragma unroll
    for (int r = 0; r < 8; ++r)
      h0[(m0 + r + 8 * half) * 64 + col] = acc[r] + bv;
  }
}

// ----- Kernel 2: s = h@Ws+bs [N,4];  hp = h@Wh+bh [N,16] --------------------
__global__ void k_proj(const float* __restrict__ h,
                       const float* __restrict__ Ws, const float* __restrict__ bs,
                       const float* __restrict__ Wh, const float* __restrict__ bh,
                       float* __restrict__ s, float* __restrict__ hp) {
  __shared__ alignas(32) _Float16 Wts[16 * 64];   // Ws^T, cols padded 4->16
  __shared__ alignas(32) _Float16 Wth[16 * 64];   // Wh^T
  const int tid = threadIdx.x;
  const int lane = tid & 31, wave = tid >> 5, half = lane >> 4;
  const int colL = lane & 15;
  const int m0 = blockIdx.x * 128 + wave * 16;
  const int rowA = m0 + colL;
  __builtin_prefetch(Ws, 0, 3);
  __builtin_prefetch(Wh, 0, 3);
  for (int i = tid; i < 16 * 64; i += 256) {
    const int col = i >> 6, k = i & 63;
    Wts[i] = (col < SDIM) ? (_Float16)Ws[k * SDIM + col] : (_Float16)0.f;
    Wth[i] = (_Float16)Wh[k * PDIM + col];
  }
  __syncthreads();
  {
    v8f acc = wmma_tile(h, 64, rowA, Wts, 64, colL, 64);
    if (colL < SDIM) {
      const float bv = bs[colL];
#pragma unroll
      for (int r = 0; r < 8; ++r)
        s[(m0 + r + 8 * half) * SDIM + colL] = acc[r] + bv;
    }
  }
  {
    v8f acc = wmma_tile(h, 64, rowA, Wth, 64, colL, 64);
    const float bv = bh[colL];
#pragma unroll
    for (int r = 0; r < 8; ++r)
      hp[(m0 + r + 8 * half) * PDIM + colL] = acc[r] + bv;
  }
}

// ----- Kernel 3: kNN (k=12) in s-space per graph + weighted mean/max agg ----
// 128 blocks x 256 thr; block owns 64 nodes of one graph; graph's s in LDS.
__global__ void k_knn(const float* __restrict__ s, const float* __restrict__ hp,
                      float* __restrict__ agg) {
  __shared__ float4 s_sh[GSZ];
  __shared__ float  mg_d[8][32 * KNN];
  __shared__ int    mg_i[8][32 * KNN];
  const int tid = threadIdx.x, lane = tid & 31, wave = tid >> 5;
  const int g0 = (blockIdx.x >> 4) * GSZ;        // 16 blocks per graph
  const int nbase = blockIdx.x * 64;

  for (int i = tid; i < GSZ; i += 256)
    s_sh[i] = ((const float4*)s)[g0 + i];
  __syncthreads();

  for (int t = 0; t < 8; ++t) {
    const int node = nbase + wave * 8 + t;
    const float4 si = s_sh[node - g0];

    // per-lane sorted 12-best over its 32 candidates (registers, unrolled)
    float bd[KNN]; int bi[KNN];
#pragma unroll
    for (int q = 0; q < KNN; ++q) { bd[q] = 3.0e38f; bi[q] = g0; }
#pragma unroll 4
    for (int c = 0; c < 32; ++c) {
      const int j = lane + 32 * c;
      const float4 sj = s_sh[j];
      const float dx = si.x - sj.x, dy = si.y - sj.y;
      const float dz = si.z - sj.z, dw = si.w - sj.w;
      float d2 = dx * dx + dy * dy + dz * dz + dw * dw;
      if (d2 < bd[KNN - 1]) {
        float cd = d2; int ci = g0 + j;
#pragma unroll
        for (int q = 0; q < KNN; ++q) {
          if (cd < bd[q]) {
            const float td = bd[q]; const int ti = bi[q];
            bd[q] = cd; bi[q] = ci; cd = td; ci = ti;
          }
        }
      }
    }
#pragma unroll
    for (int q = 0; q < KNN; ++q) {
      mg_d[wave][lane * KNN + q] = bd[q];
      mg_i[wave][lane * KNN + q] = bi[q];
    }

    // merge 32 sorted lists -> global top-12 via head-pointer wave argmin
    int ptr = 0;
    int   wn_i[KNN];
    float wn_w[KNN];
#pragma unroll
    for (int r = 0; r < KNN; ++r) {
      float bv = (ptr < KNN) ? mg_d[wave][lane * KNN + ptr] : 3.0e38f;
      int  bix = (ptr < KNN) ? mg_i[wave][lane * KNN + ptr] : g0;
      int   bl = lane;
#pragma unroll
      for (int off = 16; off >= 1; off >>= 1) {
        const float ov = __shfl_xor(bv, off);
        const int   oi = __shfl_xor(bix, off);
        const int   ol = __shfl_xor(bl, off);
        if (ov < bv || (ov == bv && ol < bl)) { bv = ov; bix = oi; bl = ol; }
      }
      if (lane == bl) ++ptr;
      wn_i[r] = bix;                    // all lanes hold winner after butterfly
      wn_w[r] = expf(-10.f * bv);       // exp(10 * neg_d2)
    }

    // aggregation: dim = lane&15 (both halves compute, lanes 0-15 store)
    const int dim = lane & 15;
    float sum = 0.f, mx = -3.0e38f;
#pragma unroll
    for (int q = 0; q < KNN; ++q) {
      const float v = hp[wn_i[q] * PDIM + dim] * wn_w[q];
      sum += v;
      mx = fmaxf(mx, v);
    }
    if (lane < 16) {
      agg[node * 32 + dim]      = sum * (1.f / 12.f);
      agg[node * 32 + 16 + dim] = mx;
    }
  }
}

// ----- Kernel 4: hn = relu([h | agg] @ Wo + bo)  (K = 96) -------------------
__global__ void k_update(const float* __restrict__ h, const float* __restrict__ agg,
                         const float* __restrict__ Wo, const float* __restrict__ bo,
                         float* __restrict__ hn) {
  __shared__ alignas(32) _Float16 Wt[64 * 96];    // Wo^T (96x64 -> [64][96])
  const int tid = threadIdx.x;
  const int lane = tid & 31, wave = tid >> 5, half = lane >> 4;
  const int colL = lane & 15;
  const int m0 = blockIdx.x * 128 + wave * 16;
  const int rowA = m0 + colL;
  __builtin_prefetch(Wo, 0, 3);
  for (int i = tid; i < 64 * 96; i += 256) {
    const int col = i / 96, k = i - col * 96;
    Wt[i] = (_Float16)Wo[k * 64 + col];
  }
  __syncthreads();
#pragma unroll
  for (int nt = 0; nt < 4; ++nt) {
    const int col = nt * 16 + colL;
    v8f acc = {};
#pragma unroll
    for (int k0 = 0; k0 < 96; k0 += 32)
      acc = wmma_acc(load_a_concat(h, agg, rowA, k0), load_b_lds(Wt, 96, col, k0), acc);
    const float bv = bo[col];
#pragma unroll
    for (int r = 0; r < 8; ++r)
      hn[(m0 + r + 8 * half) * 64 + col] = fmaxf(acc[r] + bv, 0.f);
  }
}

// ----- Kernel 5: out = softplus(relu(h[cidx] @ W1 + b1) @ w2 + b2) ----------
// 32 blocks x 256 thr over 4096 centers; wave handles 16 centers.
__global__ void k_head(const float* __restrict__ h, const int* __restrict__ cidx,
                       const float* __restrict__ W1, const float* __restrict__ b1,
                       const float* __restrict__ w2, const float* __restrict__ b2,
                       float* __restrict__ out) {
  __shared__ alignas(32) _Float16 Wt[64 * 64];    // Wo1^T
  const int tid = threadIdx.x;
  const int lane = tid & 31, wave = tid >> 5, half = lane >> 4;
  const int colL = lane & 15;
  const int m0 = blockIdx.x * 128 + wave * 16;     // center ordinal base
  __builtin_prefetch(W1, 0, 3);
  for (int i = tid; i < 64 * 64; i += 256) {
    const int col = i >> 6, k = i & 63;
    Wt[i] = (_Float16)W1[k * 64 + col];
  }
  __syncthreads();
  const int crow = cidx[m0 + colL];                // gathered A row
  float p[8];
#pragma unroll
  for (int r = 0; r < 8; ++r) p[r] = 0.f;
#pragma unroll
  for (int nt = 0; nt < 4; ++nt) {
    const int col = nt * 16 + colL;
    v8f acc = wmma_tile(h, 64, crow, Wt, 64, col, 64);
    const float bv = b1[col];
    const float wv = w2[col];
#pragma unroll
    for (int r = 0; r < 8; ++r)
      p[r] += fmaxf(acc[r] + bv, 0.f) * wv;
  }
  // reduce over the 16 lanes of each half (rows r+8*half)
#pragma unroll
  for (int off = 8; off >= 1; off >>= 1) {
#pragma unroll
    for (int r = 0; r < 8; ++r) p[r] += __shfl_xor(p[r], off);
  }
  if (colL == 0) {
    const float bb = b2[0];
#pragma unroll
    for (int r = 0; r < 8; ++r) {
      const float v = p[r] + bb;
      out[m0 + r + 8 * half] = fmaxf(v, 0.f) + log1pf(expf(-fabsf(v)));
    }
  }
}

// ---------------------------------------------------------------------------
extern "C" void kernel_launch(void* const* d_in, const int* in_sizes, int n_in,
                              void* d_out, int out_size, void* d_ws, size_t ws_size,
                              hipStream_t stream) {
  const float* x     = (const float*)d_in[0];
  const int*   cidx  = (const int*)  d_in[2];
  const float* Wi1   = (const float*)d_in[3];
  const float* bi1   = (const float*)d_in[4];
  const float* Wi2   = (const float*)d_in[5];
  const float* bi2   = (const float*)d_in[6];
  const float* c1_Ws = (const float*)d_in[7];
  const float* c1_bs = (const float*)d_in[8];
  const float* c1_Wh = (const float*)d_in[9];
  const float* c1_bh = (const float*)d_in[10];
  const float* c1_Wo = (const float*)d_in[11];
  const float* c1_bo = (const float*)d_in[12];
  const float* c2_Ws = (const float*)d_in[13];
  const float* c2_bs = (const float*)d_in[14];
  const float* c2_Wh = (const float*)d_in[15];
  const float* c2_bh = (const float*)d_in[16];
  const float* c2_Wo = (const float*)d_in[17];
  const float* c2_bo = (const float*)d_in[18];
  const float* Wo1   = (const float*)d_in[19];
  const float* bo1   = (const float*)d_in[20];
  const float* Wo2   = (const float*)d_in[21];
  const float* bo2   = (const float*)d_in[22];
  float* out = (float*)d_out;

  // workspace layout (floats): hA | hB | s | hp | agg  (~5.9 MB total)
  float* ws  = (float*)d_ws;
  float* hA  = ws;                         // 8192*64
  float* hB  = hA  + NTOT * HID;           // 8192*64
  float* s   = hB  + NTOT * HID;           // 8192*4
  float* hp  = s   + NTOT * SDIM;          // 8192*16
  float* agg = hp  + NTOT * PDIM;          // 8192*32

  k_in_mlp<<<64, 256, 0, stream>>>(x, Wi1, bi1, Wi2, bi2, hA);

  k_proj  <<<64, 256, 0, stream>>>(hA, c1_Ws, c1_bs, c1_Wh, c1_bh, s, hp);
  k_knn   <<<128, 256, 0, stream>>>(s, hp, agg);
  k_update<<<64, 256, 0, stream>>>(hA, agg, c1_Wo, c1_bo, hB);

  k_proj  <<<64, 256, 0, stream>>>(hB, c2_Ws, c2_bs, c2_Wh, c2_bh, s, hp);
  k_knn   <<<128, 256, 0, stream>>>(s, hp, agg);
  k_update<<<64, 256, 0, stream>>>(hB, agg, c2_Wo, c2_bo, hA);

  k_head  <<<32, 256, 0, stream>>>(hA, cidx, Wo1, bo1, Wo2, bo2, out);
}
